// MultiHeadAttention_79680233276252
// MI455X (gfx1250) — compile-verified, matrix operand order checked by reading, not run on hardware
//
#include <hip/hip_runtime.h>
#include <hip/hip_bf16.h>
#include <math.h>

#define S_LEN 4096
#define DIMSZ 1024
#define NH    16
#define HD    64

typedef __bf16 bf16_t;
typedef bf16_t v8bf  __attribute__((ext_vector_type(8)));
typedef bf16_t v16bf __attribute__((ext_vector_type(16)));
typedef float  v8f   __attribute__((ext_vector_type(8)));
typedef int    v4i_vs __attribute__((vector_size(16)));   // matches builtin param

__device__ __forceinline__ unsigned short f32_to_bf16(float f) {
    unsigned int u = __float_as_uint(f);
    u += 0x7FFFu + ((u >> 16) & 1u);   // round-to-nearest-even
    return (unsigned short)(u >> 16);
}
__device__ __forceinline__ float bf16_to_f32(unsigned short h) {
    return __uint_as_float(((unsigned int)h) << 16);
}
// Build a 16-element bf16 WMMA operand from two 16-byte chunks (global or LDS).
__device__ __forceinline__ v16bf load_ab(const unsigned short* p0, const unsigned short* p1) {
    v8bf lo = *(const v8bf*)p0;
    v8bf hi = *(const v8bf*)p1;
    return __builtin_shufflevector(lo, hi, 0,1,2,3,4,5,6,7,8,9,10,11,12,13,14,15);
}
#define WMMA_BF16(a, b, c) \
    __builtin_amdgcn_wmma_f32_16x16x32_bf16(false, (a), false, (b), (short)0, (c), false, false)

// ---- gfx1250 async global->LDS copy (ASYNCcnt), with safe fallback --------
#if __has_builtin(__builtin_amdgcn_global_load_async_to_lds_b128)
#define HAS_ASYNC_LDS 1
#else
#define HAS_ASYNC_LDS 0
#endif

__device__ __forceinline__ void async_copy16(const unsigned short* g, unsigned short* l) {
#if HAS_ASYNC_LDS
    __builtin_amdgcn_global_load_async_to_lds_b128(
        (__attribute__((address_space(1))) v4i_vs*)g,
        (__attribute__((address_space(3))) v4i_vs*)l, 0, 0);
#else
    *(v8bf*)l = *(const v8bf*)g;     // synchronous staging fallback
#endif
}

template <int N>
__device__ __forceinline__ void wait_asynccnt() {
#if HAS_ASYNC_LDS
#if __has_builtin(__builtin_amdgcn_s_wait_asynccnt)
    __builtin_amdgcn_s_wait_asynccnt(N);
#else
    asm volatile("s_wait_asynccnt %0" :: "i"(N) : "memory");
#endif
#endif
}

// ---------------------------------------------------------------------------
// fp32 -> bf16 cast
// ---------------------------------------------------------------------------
__global__ void f2bf_kernel(const float* __restrict__ in, unsigned short* __restrict__ out, int n) {
    int i = blockIdx.x * blockDim.x + threadIdx.x;
    if (i < n) out[i] = f32_to_bf16(in[i]);
}

// ---------------------------------------------------------------------------
// C = A[M,K] * B[N,K]^T, bf16 in, f32 accumulate (v_wmma_f32_16x16x32_bf16).
// Block tile 256x64 (8 waves x 32 rows), B k-slice (4KB) async-staged to LDS
// once per block per k-step, double buffered; each wave does 8 WMMAs/k-step.
// mode 0: Cf[m*N+n] fp32            mode 1: bf16 [head][seq][64]
// mode 2: bf16 [head][64][seq] (transposed V)
// grid = (M/256, N/64), block = 256
// ---------------------------------------------------------------------------
__global__ __launch_bounds__(256)
void gemm_bf16_wmma(const unsigned short* __restrict__ A,
                    const unsigned short* __restrict__ B,
                    float* __restrict__ Cf,
                    unsigned short* __restrict__ Cb,
                    int M, int N, int K, int mode) {
    __shared__ __align__(16) unsigned short Blds[2][2048];   // 2 x 4KB

    const int tid  = threadIdx.x;
    const int lane = tid & 31;
    const int wave = tid >> 5;
    const int col  = lane & 15;
    const int hi   = lane >> 4;
    const int m0   = blockIdx.x * 256 + wave * 32;
    const int n0   = blockIdx.y * 64;

    // producer: one 16B chunk per thread per k-step
    // tid = t*64 + hf*32 + l  ->  B row n0+t*16+(l&15), k offset hf*16+(l>>4)*8
    const int pt  = tid >> 6;
    const int phf = (tid >> 5) & 1;
    const int pl  = tid & 31;
    const unsigned short* bsrc = B + (size_t)(n0 + pt * 16 + (pl & 15)) * K
                                   + phf * 16 + (pl >> 4) * 8;

    const unsigned short* ar0 = A + (size_t)(m0 + col) * K;
    const unsigned short* ar1 = A + (size_t)(m0 + 16 + col) * K;

    v8f acc0[4] = {};
    v8f acc1[4] = {};

    async_copy16(bsrc, &Blds[0][tid * 8]);          // prologue stage k0=0

    for (int k0 = 0; k0 < K; k0 += 32) {
        const int cur = (k0 >> 5) & 1;
        __syncthreads();                            // WAR: done reading Blds[cur^1]
        if (k0 + 32 < K) {
            async_copy16(bsrc + k0 + 32, &Blds[cur ^ 1][tid * 8]);
            wait_asynccnt<1>();                     // copies into Blds[cur] landed
        } else {
            wait_asynccnt<0>();
        }
        __syncthreads();                            // visible to all waves

        v16bf a0 = load_ab(ar0 + k0 + hi * 8, ar0 + k0 + 16 + hi * 8);
        v16bf a1 = load_ab(ar1 + k0 + hi * 8, ar1 + k0 + 16 + hi * 8);
        const unsigned short* Bl = &Blds[cur][0];
#pragma unroll
        for (int t = 0; t < 4; ++t) {
            v16bf b = load_ab(Bl + (t * 64 + lane) * 8, Bl + (t * 64 + 32 + lane) * 8);
            acc0[t] = WMMA_BF16(a0, b, acc0[t]);
            acc1[t] = WMMA_BF16(a1, b, acc1[t]);
        }
    }

#pragma unroll
    for (int t = 0; t < 4; ++t) {
#pragma unroll
        for (int v = 0; v < 8; ++v) {
            int n = n0 + t * 16 + col;
#pragma unroll
            for (int half = 0; half < 2; ++half) {
                int m = m0 + half * 16 + hi * 8 + v;
                float val = half ? acc1[t][v] : acc0[t][v];
                if (mode == 0) {
                    Cf[(size_t)m * N + n] = val;
                } else {
                    int h = n >> 6, d = n & 63;
                    if (mode == 1) Cb[((size_t)h * M + m) * 64 + d] = f32_to_bf16(val);
                    else           Cb[((size_t)h * 64 + d) * M + m] = f32_to_bf16(val);
                }
            }
        }
    }
}

// ---------------------------------------------------------------------------
// In-place interleaved RoPE on Q and K ([head][seq][64] bf16).
// ---------------------------------------------------------------------------
__global__ void rope_kernel(unsigned short* __restrict__ Qh, unsigned short* __restrict__ Kh) {
    int idx = blockIdx.x * blockDim.x + threadIdx.x;
    if (idx >= NH * S_LEN * 32) return;
    int pair = idx & 31;
    int s    = (idx >> 5) & (S_LEN - 1);
    int h    = idx >> 17;
    float inv = __powf(10000.0f, -(float)pair / 32.0f);
    float ang = (float)s * inv;
    float c = cosf(ang), sn = sinf(ang);
    size_t off = (((size_t)h * S_LEN + s) * HD) + 2 * pair;
    {
        float xe = bf16_to_f32(Qh[off]), xo = bf16_to_f32(Qh[off + 1]);
        Qh[off]     = f32_to_bf16(xe * c - xo * sn);
        Qh[off + 1] = f32_to_bf16(xo * c + xe * sn);
    }
    {
        float xe = bf16_to_f32(Kh[off]), xo = bf16_to_f32(Kh[off + 1]);
        Kh[off]     = f32_to_bf16(xe * c - xo * sn);
        Kh[off + 1] = f32_to_bf16(xo * c + xe * sn);
    }
}

// ---------------------------------------------------------------------------
// Causal flash attention, one wave per (head, 16-query tile).
// K and V tiles for the NEXT 32-key block are async-staged into an LDS double
// buffer while the current block computes (4 QK^T WMMAs + online softmax +
// 4 PV WMMAs). Async loads complete in order -> s_wait_asynccnt 16 after
// issuing the next 16 copies guarantees the current 16 have landed.
// ---------------------------------------------------------------------------
__device__ __forceinline__ void stage_kv(const unsigned short* __restrict__ Kh,
                                         const unsigned short* __restrict__ Vt,
                                         int h, int kb, int lane,
                                         unsigned short* buf) {
    const int col = lane & 15, hi = lane >> 4;
    const unsigned short* kr0 = Kh + ((size_t)h * S_LEN + kb + col) * HD + hi * 8;
    const unsigned short* kr1 = kr0 + 16 * HD;
    async_copy16(kr0,      buf + (0 * 32 + lane) * 8);
    async_copy16(kr0 + 16, buf + (1 * 32 + lane) * 8);
    async_copy16(kr0 + 32, buf + (2 * 32 + lane) * 8);
    async_copy16(kr0 + 48, buf + (3 * 32 + lane) * 8);
    async_copy16(kr1,      buf + (4 * 32 + lane) * 8);
    async_copy16(kr1 + 16, buf + (5 * 32 + lane) * 8);
    async_copy16(kr1 + 32, buf + (6 * 32 + lane) * 8);
    async_copy16(kr1 + 48, buf + (7 * 32 + lane) * 8);
#pragma unroll
    for (int dt = 0; dt < 4; ++dt) {
        const unsigned short* vr = Vt + ((size_t)h * 64 + dt * 16 + col) * S_LEN + kb + hi * 8;
        async_copy16(vr,      buf + ((8 + 2 * dt) * 32 + lane) * 8);
        async_copy16(vr + 16, buf + ((9 + 2 * dt) * 32 + lane) * 8);
    }
}

__global__ __launch_bounds__(32)
void attn_wmma(const unsigned short* __restrict__ Qh,
               const unsigned short* __restrict__ Kh,
               const unsigned short* __restrict__ Vt,
               unsigned short* __restrict__ AO) {
    const int lane = threadIdx.x & 31;
    const int col  = lane & 15;
    const int hi   = lane >> 4;
    const int h    = blockIdx.y;
    const int q0   = blockIdx.x * 16;

    __shared__ __align__(16) unsigned short Plds[16 * 32];
    __shared__ __align__(16) unsigned short KV[2][16 * 32 * 8];   // 2 x 8KB

    const unsigned short* q_row = Qh + ((size_t)h * S_LEN + q0 + col) * HD;
    v16bf aq0 = load_ab(q_row + hi * 8,      q_row + 16 + hi * 8);
    v16bf aq1 = load_ab(q_row + 32 + hi * 8, q_row + 48 + hi * 8);

    v8f acc[4] = {};
    float mrow[8], lrow[8];
#pragma unroll
    for (int v = 0; v < 8; ++v) { mrow[v] = -INFINITY; lrow[v] = 0.0f; }

    const float scale = 0.125f;           // 1/sqrt(64)
    const int kend = q0 + 16;             // causal: keys <= q0+15

    stage_kv(Kh, Vt, h, 0, lane, &KV[0][0]);   // prologue

    for (int kb = 0; kb < kend; kb += 32) {
        const int cur = (kb >> 5) & 1;
        if (kb + 32 < kend) {
            stage_kv(Kh, Vt, h, kb + 32, lane, &KV[cur ^ 1][0]);
            wait_asynccnt<16>();          // current block's 16 copies landed
        } else {
            wait_asynccnt<0>();
        }
        const unsigned short* Kb = &KV[cur][0];

        // ---- scores: 16 queries x 32 keys (from LDS) --------------------
        v8f s0 = {}, s1 = {};
        {
            v16bf b0 = load_ab(Kb + (0 * 32 + lane) * 8, Kb + (1 * 32 + lane) * 8);
            v16bf b1 = load_ab(Kb + (2 * 32 + lane) * 8, Kb + (3 * 32 + lane) * 8);
            s0 = WMMA_BF16(aq0, b0, s0);
            s0 = WMMA_BF16(aq1, b1, s0);
            v16bf c0 = load_ab(Kb + (4 * 32 + lane) * 8, Kb + (5 * 32 + lane) * 8);
            v16bf c1 = load_ab(Kb + (6 * 32 + lane) * 8, Kb + (7 * 32 + lane) * 8);
            s1 = WMMA_BF16(aq0, c0, s1);
            s1 = WMMA_BF16(aq1, c1, s1);
        }

        // ---- online softmax (C layout: row = hi*8+v, col = lane&15) -----
        const int key0 = kb + col;
        const int key1 = kb + 16 + col;
#pragma unroll
        for (int v = 0; v < 8; ++v) {
            int q = q0 + hi * 8 + v;
            float s0v = (key0 <= q) ? s0[v] * scale : -INFINITY;
            float s1v = (key1 <= q) ? s1[v] * scale : -INFINITY;
            float mx = fmaxf(s0v, s1v);
            mx = fmaxf(mx, __shfl_xor(mx, 1));
            mx = fmaxf(mx, __shfl_xor(mx, 2));
            mx = fmaxf(mx, __shfl_xor(mx, 4));
            mx = fmaxf(mx, __shfl_xor(mx, 8));
            float mnew  = fmaxf(mrow[v], mx);
            float alpha = __expf(mrow[v] - mnew);
            float p0 = __expf(s0v - mnew);
            float p1 = __expf(s1v - mnew);
            float rs = p0 + p1;
            rs += __shfl_xor(rs, 1);
            rs += __shfl_xor(rs, 2);
            rs += __shfl_xor(rs, 4);
            rs += __shfl_xor(rs, 8);
            lrow[v] = lrow[v] * alpha + rs;
            mrow[v] = mnew;
            acc[0][v] *= alpha; acc[1][v] *= alpha; acc[2][v] *= alpha; acc[3][v] *= alpha;
            int m = hi * 8 + v;
            Plds[m * 32 + col]      = f32_to_bf16(p0);
            Plds[m * 32 + 16 + col] = f32_to_bf16(p1);
        }
        __syncthreads();
        v16bf ap = load_ab(&Plds[col * 32 + hi * 8], &Plds[col * 32 + 16 + hi * 8]);
        __syncthreads();

        // ---- P(16x32 keys) * V(32 keys x 64 d), V from LDS --------------
#pragma unroll
        for (int dt = 0; dt < 4; ++dt) {
            v16bf bv = load_ab(Kb + ((8 + 2 * dt) * 32 + lane) * 8,
                               Kb + ((9 + 2 * dt) * 32 + lane) * 8);
            acc[dt] = WMMA_BF16(ap, bv, acc[dt]);
        }
    }

#pragma unroll
    for (int dt = 0; dt < 4; ++dt) {
#pragma unroll
        for (int v = 0; v < 8; ++v) {
            int m = q0 + hi * 8 + v;
            int d = dt * 16 + col;
            float o = acc[dt][v] / lrow[v];
            AO[(size_t)m * DIMSZ + h * HD + d] = f32_to_bf16(o);
        }
    }
}

// ---------------------------------------------------------------------------
extern "C" void kernel_launch(void* const* d_in, const int* in_sizes, int n_in,
                              void* d_out, int out_size, void* d_ws, size_t ws_size,
                              hipStream_t stream) {
    const float* x  = (const float*)d_in[0];
    const float* Wq = (const float*)d_in[1];
    const float* Wk = (const float*)d_in[2];
    const float* Wv = (const float*)d_in[3];
    const float* Wo = (const float*)d_in[4];
    float* out = (float*)d_out;

    unsigned short* ws = (unsigned short*)d_ws;
    const size_t SD = (size_t)S_LEN * DIMSZ;  // 4 Mi elems
    const size_t WW = (size_t)DIMSZ * DIMSZ;  // 1 Mi elems
    unsigned short* xb  = ws;
    unsigned short* Wqb = xb  + SD;
    unsigned short* Wkb = Wqb + WW;
    unsigned short* Wvb = Wkb + WW;
    unsigned short* Wob = Wvb + WW;
    unsigned short* Qh  = Wob + WW;           // [head][seq][64]
    unsigned short* Kh  = Qh  + SD;           // [head][seq][64]
    unsigned short* Vt  = Kh  + SD;           // [head][64][seq]
    unsigned short* AO  = Vt  + SD;           // [seq][dim]

    f2bf_kernel<<<dim3((unsigned)((SD + 255) / 256)), 256, 0, stream>>>(x,  xb,  (int)SD);
    f2bf_kernel<<<dim3((unsigned)((WW + 255) / 256)), 256, 0, stream>>>(Wq, Wqb, (int)WW);
    f2bf_kernel<<<dim3((unsigned)((WW + 255) / 256)), 256, 0, stream>>>(Wk, Wkb, (int)WW);
    f2bf_kernel<<<dim3((unsigned)((WW + 255) / 256)), 256, 0, stream>>>(Wv, Wvb, (int)WW);
    f2bf_kernel<<<dim3((unsigned)((WW + 255) / 256)), 256, 0, stream>>>(Wo, Wob, (int)WW);

    dim3 ggrid(S_LEN / 256, DIMSZ / 64);      // (16, 16)
    gemm_bf16_wmma<<<ggrid, 256, 0, stream>>>(xb, Wqb, nullptr, Qh, S_LEN, DIMSZ, DIMSZ, 1);
    gemm_bf16_wmma<<<ggrid, 256, 0, stream>>>(xb, Wkb, nullptr, Kh, S_LEN, DIMSZ, DIMSZ, 1);
    gemm_bf16_wmma<<<ggrid, 256, 0, stream>>>(xb, Wvb, nullptr, Vt, S_LEN, DIMSZ, DIMSZ, 2);

    int nrope = NH * S_LEN * 32;
    rope_kernel<<<(nrope + 255) / 256, 256, 0, stream>>>(Qh, Kh);

    attn_wmma<<<dim3(S_LEN / 16, NH), 32, 0, stream>>>(Qh, Kh, Vt, AO);

    gemm_bf16_wmma<<<ggrid, 256, 0, stream>>>(AO, Wob, out, nullptr, S_LEN, DIMSZ, DIMSZ, 0);
}
